// MODEL_87943750353422
// MI455X (gfx1250) — matrix operand order repaired and checked by reading, not compiled, not run on hardware
//
#include <hip/hip_runtime.h>
#include <math.h>

// ---------------------------------------------------------------------------
// DKVMN-style model on MI455X (gfx1250, wave32, WMMA).
// All GEMMs use v_wmma_f32_16x16x32_f16 (f16 in, f32 accumulate).
// The recurrent scan is hoisted: erase/add/w are batched GEMMs over B*S rows;
// the scan itself keeps Mv columns in registers (64 f32 per thread).
// ---------------------------------------------------------------------------

typedef __attribute__((ext_vector_type(16))) _Float16 v16h;
typedef __attribute__((ext_vector_type(8)))  float    v8f;

#define WMMA16(a, b, c) \
  __builtin_amdgcn_wmma_f32_16x16x32_f16(false, (a), false, (b), (short)0, (c), false, false)

// A-matrix (16x32 f16) K index for fragment element j, lane-half hi:
// lanes 0-15: j=0..7 -> K0..7,  j=8..15 -> K16..23
// lanes16-31: j=0..7 -> K8..15, j=8..15 -> K24..31
__device__ __forceinline__ int akpos(int j, int hi) {
  return (j & 7) + (hi ? 8 : 0) + ((j & 8) ? 16 : 0);
}
// B-matrix (32x16 f16) K index: same as A (lanes 0-15: K0..7,K16..23; lanes 16-31: K8..15,K24..31)
__device__ __forceinline__ int bkpos(int j, int hi) { return (j & 7) + (hi ? 8 : 0) + ((j & 8) ? 16 : 0); }

static constexpr int Bn  = 128;
static constexpr int Sn  = 1000;
static constexpr int BS  = Bn * Sn;       // 128000
static constexpr int QD  = 64;
static constexpr int QAD = 256;
static constexpr int Mn  = 64;
static constexpr int VD  = 256;
static constexpr int FC  = 512;
static constexpr int CAT = VD + QD;       // 320

// ---------------------------------------------------------------------------
__global__ void f32_to_f16_kernel(const float* __restrict__ src,
                                  _Float16* __restrict__ dst, int n) {
  int i = blockIdx.x * 256 + threadIdx.x;
  if (i < n) dst[i] = (_Float16)src[i];
}

// ---------------------------------------------------------------------------
// w = softmax(qe @ key_mem^T)  and  inp = tanh(qe @ input_W^T + input_b)
// One wave handles 16 rows (bs). K=64 -> 2 k-tiles. N=64 -> 4 n-tiles.
__global__ __launch_bounds__(32)
void w_inp_kernel(const int* __restrict__ qd, const float* __restrict__ qtab,
                  const _Float16* __restrict__ kW, const _Float16* __restrict__ iW,
                  const float* __restrict__ ib,
                  float* __restrict__ w_all, _Float16* __restrict__ feat) {
  const int bs0  = blockIdx.x * 16;
  const int lane = threadIdx.x & 31;
  const int hi   = lane >> 4;
  const int row  = lane & 15;

  const float* arow = qtab + (size_t)qd[bs0 + row] * QD;
  v16h a0, a1;
#pragma unroll
  for (int j = 0; j < 16; ++j) {
    a0[j] = (_Float16)arow[akpos(j, hi)];
    a1[j] = (_Float16)arow[32 + akpos(j, hi)];
  }

  // ---- attention logits over key_mem, then row softmax (64 cols)
  float cw[4][8];
#pragma unroll
  for (int nt = 0; nt < 4; ++nt) {
    const _Float16* wrow = kW + (size_t)(nt * 16 + row) * QD;
    v16h b0, b1;
#pragma unroll
    for (int j = 0; j < 16; ++j) {
      b0[j] = wrow[bkpos(j, hi)];
      b1[j] = wrow[32 + bkpos(j, hi)];
    }
    v8f c = {};
    c = WMMA16(a0, b0, c);
    c = WMMA16(a1, b1, c);
#pragma unroll
    for (int v = 0; v < 8; ++v) cw[nt][v] = c[v];
  }
#pragma unroll
  for (int v = 0; v < 8; ++v) {
    float mx = fmaxf(fmaxf(cw[0][v], cw[1][v]), fmaxf(cw[2][v], cw[3][v]));
#pragma unroll
    for (int msk = 1; msk < 16; msk <<= 1) mx = fmaxf(mx, __shfl_xor(mx, msk, 32));
    float s = 0.f;
#pragma unroll
    for (int nt = 0; nt < 4; ++nt) { cw[nt][v] = __expf(cw[nt][v] - mx); s += cw[nt][v]; }
#pragma unroll
    for (int msk = 1; msk < 16; msk <<= 1) s += __shfl_xor(s, msk, 32);
    const float inv = 1.0f / s;
    const int m = v + hi * 8;
#pragma unroll
    for (int nt = 0; nt < 4; ++nt)
      w_all[(size_t)(bs0 + m) * Mn + nt * 16 + row] = cw[nt][v] * inv;
  }

  // ---- inp = tanh(qe @ input_W^T + b) -> feat[:, 256:320]
#pragma unroll
  for (int nt = 0; nt < 4; ++nt) {
    const _Float16* wrow = iW + (size_t)(nt * 16 + row) * QD;
    v16h b0, b1;
#pragma unroll
    for (int j = 0; j < 16; ++j) {
      b0[j] = wrow[bkpos(j, hi)];
      b1[j] = wrow[32 + bkpos(j, hi)];
    }
    v8f c = {};
    c = WMMA16(a0, b0, c);
    c = WMMA16(a1, b1, c);
    const int col  = nt * 16 + row;
    const float bv = ib[col];
#pragma unroll
    for (int v = 0; v < 8; ++v) {
      const int m = v + hi * 8;
      feat[(size_t)(bs0 + m) * CAT + VD + col] = (_Float16)tanhf(c[v] + bv);
    }
  }
}

// ---------------------------------------------------------------------------
// erase = sigmoid(qae @ eW^T + eb), add = tanh(qae @ aW^T + ab)
// One wave: 16 rows, K=256 (8 k-tiles, A cached), N=256 (16 n-tiles), 2 passes.
__global__ __launch_bounds__(32)
void erase_add_kernel(const int* __restrict__ qad, const float* __restrict__ qatab,
                      const _Float16* __restrict__ eW, const float* __restrict__ eb,
                      const _Float16* __restrict__ aW, const float* __restrict__ ab,
                      _Float16* __restrict__ eout, _Float16* __restrict__ aout) {
  const int bs0  = blockIdx.x * 16;
  const int lane = threadIdx.x & 31;
  const int hi   = lane >> 4;
  const int row  = lane & 15;

  const float* arow = qatab + (size_t)qad[bs0 + row] * QAD;
  v16h afr[8];
#pragma unroll
  for (int kt = 0; kt < 8; ++kt)
#pragma unroll
    for (int j = 0; j < 16; ++j)
      afr[kt][j] = (_Float16)arow[kt * 32 + akpos(j, hi)];

#pragma unroll
  for (int pass = 0; pass < 2; ++pass) {
    const _Float16* W    = pass ? aW : eW;
    const float*    bias = pass ? ab : eb;
    _Float16*       out  = pass ? aout : eout;
    for (int nt = 0; nt < 16; ++nt) {
      v8f c = {};
#pragma unroll
      for (int kt = 0; kt < 8; ++kt) {
        const _Float16* wrow = W + (size_t)(nt * 16 + row) * QAD + kt * 32;
        v16h b;
#pragma unroll
        for (int j = 0; j < 16; ++j) b[j] = wrow[bkpos(j, hi)];
        c = WMMA16(afr[kt], b, c);
      }
      const int col  = nt * 16 + row;
      const float bv = bias[col];
#pragma unroll
      for (int v = 0; v < 8; ++v) {
        const float f = c[v] + bv;
        const float r = pass ? tanhf(f) : (1.0f / (1.0f + __expf(-f)));
        out[(size_t)(bs0 + v + hi * 8) * VD + col] = (_Float16)r;
      }
    }
  }
}

// ---------------------------------------------------------------------------
// Sequential memory scan. One block per batch element; thread t owns Mv
// column t (64 f32 in registers). read uses old Mv, then rank-1 update.
__global__ __launch_bounds__(256)
void scan_kernel(const float* __restrict__ w_all, const _Float16* __restrict__ eh,
                 const _Float16* __restrict__ ah, const float* __restrict__ init_value,
                 _Float16* __restrict__ feat) {
  const int b = blockIdx.x;
  const int t = threadIdx.x;  // 0..255 (value column)
  __shared__ float wbuf[Mn];

  float mv[Mn];
#pragma unroll
  for (int m = 0; m < Mn; ++m) mv[m] = init_value[m * VD + t];

  for (int s = 0; s < Sn; ++s) {
    const size_t bs = (size_t)b * Sn + s;
    if (t < Mn) wbuf[t] = w_all[bs * Mn + t];
    __syncthreads();
    const float e = (float)eh[bs * VD + t];
    const float a = (float)ah[bs * VD + t];
    float racc = 0.f;
#pragma unroll
    for (int m = 0; m < Mn; ++m) {
      const float wm = wbuf[m];
      racc += wm * mv[m];                        // read with OLD Mv
      mv[m] = mv[m] + wm * (a - e * mv[m]);      // Mv*(1-w*e) + w*a
    }
    feat[bs * CAT + t] = (_Float16)racc;
    __syncthreads();
  }
}

// ---------------------------------------------------------------------------
// h = tanh(feat @ read_W^T + rb); logits = h @ pred_W^T + pb, fused (no h buffer).
// One wave: 16 rows, K=320 (10 k-tiles cached), N=512 (32 n-tiles reduced on the fly).
__global__ __launch_bounds__(32)
void head_kernel(const _Float16* __restrict__ feat, const _Float16* __restrict__ rW,
                 const float* __restrict__ rb, const float* __restrict__ pW,
                 const float* __restrict__ pb, float* __restrict__ logits) {
  const int bs0  = blockIdx.x * 16;
  const int lane = threadIdx.x & 31;
  const int hi   = lane >> 4;
  const int row  = lane & 15;

  const _Float16* arow = feat + (size_t)(bs0 + row) * CAT;
  v16h afr[10];
#pragma unroll
  for (int kt = 0; kt < 10; ++kt)
#pragma unroll
    for (int j = 0; j < 16; ++j)
      afr[kt][j] = arow[kt * 32 + akpos(j, hi)];

  float acc[8] = {0.f, 0.f, 0.f, 0.f, 0.f, 0.f, 0.f, 0.f};
  for (int nt = 0; nt < 32; ++nt) {
    v8f c = {};
#pragma unroll
    for (int kt = 0; kt < 10; ++kt) {
      const _Float16* wrow = rW + (size_t)(nt * 16 + row) * CAT + kt * 32;
      v16h b;
#pragma unroll
      for (int j = 0; j < 16; ++j) b[j] = wrow[bkpos(j, hi)];
      c = WMMA16(afr[kt], b, c);
    }
    const int col  = nt * 16 + row;
    const float bv = rb[col];
    const float pw = pW[col];
#pragma unroll
    for (int v = 0; v < 8; ++v) acc[v] += tanhf(c[v] + bv) * pw;
  }
#pragma unroll
  for (int v = 0; v < 8; ++v) {
#pragma unroll
    for (int msk = 1; msk < 16; msk <<= 1) acc[v] += __shfl_xor(acc[v], msk, 32);
  }
  if (row == 0) {
#pragma unroll
    for (int v = 0; v < 8; ++v)
      logits[bs0 + v + hi * 8] = acc[v] + pb[0];
  }
}

// ---------------------------------------------------------------------------
// Masked BCE: per-block partials (deterministic), plus sigmoid/target outputs.
__global__ __launch_bounds__(256)
void loss_partial_kernel(const float* __restrict__ logits, const float* __restrict__ target,
                         const int* __restrict__ qd, float* __restrict__ out,
                         float* __restrict__ part, float* __restrict__ cnt) {
  __shared__ float sred[256], scnt[256];
  const int tx = threadIdx.x;
  const int i = blockIdx.x * 256 + tx;
  float v = 0.f, c = 0.f;
  if (i < BS) {
    const float l = logits[i];
    out[1 + i] = 1.0f / (1.0f + __expf(-l));
    const float tg = target[i];
    out[1 + BS + i] = tg;
    if (qd[i] > 0) {
      v = fmaxf(l, 0.f) - l * tg + log1pf(__expf(-fabsf(l)));
      c = 1.f;
    }
  }
  sred[tx] = v; scnt[tx] = c;
  __syncthreads();
  for (int st = 128; st > 0; st >>= 1) {
    if (tx < st) { sred[tx] += sred[tx + st]; scnt[tx] += scnt[tx + st]; }
    __syncthreads();
  }
  if (tx == 0) { part[blockIdx.x] = sred[0]; cnt[blockIdx.x] = scnt[0]; }
}

__global__ __launch_bounds__(512)
void loss_final_kernel(const float* __restrict__ part, const float* __restrict__ cnt,
                       int nb, float* __restrict__ out) {
  __shared__ float s1[512], s2[512];
  const int tx = threadIdx.x;
  float v = 0.f, c = 0.f;
  for (int i = tx; i < nb; i += 512) { v += part[i]; c += cnt[i]; }
  s1[tx] = v; s2[tx] = c;
  __syncthreads();
  for (int st = 256; st > 0; st >>= 1) {
    if (tx < st) { s1[tx] += s1[tx + st]; s2[tx] += s2[tx + st]; }
    __syncthreads();
  }
  if (tx == 0) out[0] = s1[0] / fmaxf(s2[0], 1.0f);
}

// ---------------------------------------------------------------------------
extern "C" void kernel_launch(void* const* d_in, const int* in_sizes, int n_in,
                              void* d_out, int out_size, void* d_ws, size_t ws_size,
                              hipStream_t stream) {
  const int*   q_data   = (const int*)  d_in[0];
  const int*   qa_data  = (const int*)  d_in[1];
  const float* target   = (const float*)d_in[2];
  const float* q_table  = (const float*)d_in[6];
  const float* qa_table = (const float*)d_in[7];
  const float* key_mem  = (const float*)d_in[8];
  const float* init_val = (const float*)d_in[9];
  const float* erase_W  = (const float*)d_in[10];
  const float* erase_b  = (const float*)d_in[11];
  const float* add_W    = (const float*)d_in[12];
  const float* add_b    = (const float*)d_in[13];
  const float* input_W  = (const float*)d_in[14];
  const float* input_b  = (const float*)d_in[15];
  const float* read_W   = (const float*)d_in[16];
  const float* read_b   = (const float*)d_in[17];
  const float* pred_W   = (const float*)d_in[18];
  const float* pred_b   = (const float*)d_in[19];
  float* out = (float*)d_out;

  // workspace carve-up (256-byte aligned chunks)
  char* p = (char*)d_ws;
  auto take = [&](size_t bytes) {
    char* r = p;
    p += (bytes + 255) & ~(size_t)255;
    return r;
  };
  float*    w_all   = (float*)   take((size_t)BS * Mn * 4);   // 32.8 MB
  _Float16* erase_h = (_Float16*)take((size_t)BS * VD * 2);   // 65.5 MB
  _Float16* add_h   = (_Float16*)take((size_t)BS * VD * 2);   // 65.5 MB
  _Float16* feat_h  = (_Float16*)take((size_t)BS * CAT * 2);  // 81.9 MB
  float*    logits  = (float*)   take((size_t)BS * 4);        // 0.5 MB
  _Float16* eW_h    = (_Float16*)take((size_t)VD * QAD * 2);
  _Float16* aW_h    = (_Float16*)take((size_t)VD * QAD * 2);
  _Float16* rW_h    = (_Float16*)take((size_t)FC * CAT * 2);
  _Float16* iW_h    = (_Float16*)take((size_t)QD * QD * 2);
  _Float16* kW_h    = (_Float16*)take((size_t)Mn * QD * 2);
  float*    part    = (float*)   take(512 * 4);
  float*    cntp    = (float*)   take(512 * 4);

  // 1) convert weights to f16 once per call
  auto conv = [&](const float* s, _Float16* d, int n) {
    f32_to_f16_kernel<<<(n + 255) / 256, 256, 0, stream>>>(s, d, n);
  };
  conv(erase_W, eW_h, VD * QAD);
  conv(add_W,   aW_h, VD * QAD);
  conv(read_W,  rW_h, FC * CAT);
  conv(input_W, iW_h, QD * QD);
  conv(key_mem, kW_h, Mn * QD);

  // 2) batched attention weights + input branch (WMMA)
  w_inp_kernel<<<BS / 16, 32, 0, stream>>>(q_data, q_table, kW_h, iW_h, input_b,
                                           w_all, feat_h);
  // 3) batched erase / add gates (WMMA)
  erase_add_kernel<<<BS / 16, 32, 0, stream>>>(qa_data, qa_table, eW_h, erase_b,
                                               aW_h, add_b, erase_h, add_h);
  // 4) recurrent memory scan (register-resident Mv)
  scan_kernel<<<Bn, 256, 0, stream>>>(w_all, erase_h, add_h, init_val, feat_h);
  // 5) head GEMM fused with prediction dot product (WMMA)
  head_kernel<<<BS / 16, 32, 0, stream>>>(feat_h, rW_h, read_b, pred_W, pred_b, logits);
  // 6) masked BCE (two-stage deterministic reduction) + outputs
  loss_partial_kernel<<<(BS + 255) / 256, 256, 0, stream>>>(logits, target, q_data,
                                                            out, part, cntp);
  loss_final_kernel<<<1, 512, 0, stream>>>(part, cntp, (BS + 255) / 256, out);
}
